// NMS3dAndComposeA_1949915152762
// MI455X (gfx1250) — matrix operand order repaired and checked by reading, not compiled, not run on hardware
//
#include <hip/hip_runtime.h>
#include <math.h>
#include <stdint.h>

// ---------------- tunables / constants ----------------
#define TW 32              // tile width  (block x)
#define TH 8               // tile height (block y)
#define TPADL 4            // left pad so B128 async loads stay 16B-aligned
#define TCOLS (TW + 2 * TPADL)   // 40 columns: covers [gx0-4, gx0+36)
#define TROWS (TH + 2)           // 10 rows:    covers [gy0-1, gy0+9)
#define TSIZE (TROWS * TCOLS)
#define NBINS 8192
#define CAP 8192           // candidate capacity (power of 2, fits 64KB LDS sort)
#define NMS_EPS_C 1e-5f
#define SAM_EPS_C 1e-8f
#define BORDER_C 3

typedef __attribute__((ext_vector_type(16))) _Float16 v16h;
typedef __attribute__((ext_vector_type(8)))  float    v8f;

// ---------------- async tile staging (gfx1250 ASYNCcnt path) ----------------
// Interior blocks: B128 async global->LDS copies (16B aligned, 100 ops/tile).
// Boundary blocks: clamped B32 async copies (uniform per-block branch).
// Low 32 bits of a flat shared-pointer are the LDS byte address (ISA 10.2).
__device__ __forceinline__ void issue_tile_async(float* tile, const float* __restrict__ src,
                                                 int bx, int by, int H, int W, int tid, bool fast) {
  int gx0 = bx * TW;
  int gy0 = by * TH;
  if (fast) {
    for (int i = tid; i < TROWS * (TCOLS / 4); i += TW * TH) {   // 100 B128 ops
      int r = i / (TCOLS / 4);
      int seg = i % (TCOLS / 4);
      unsigned goff = (unsigned)((((size_t)(gy0 - 1 + r)) * (size_t)W +
                                  (size_t)(gx0 - TPADL + seg * 4)) * 4u);
      unsigned loff = (unsigned)(uintptr_t)(&tile[r * TCOLS + seg * 4]);
      asm volatile("global_load_async_to_lds_b128 %0, %1, %2"
                   :: "v"(loff), "v"(goff), "s"(src) : "memory");
    }
  } else {
    for (int i = tid; i < TSIZE; i += TW * TH) {                 // 400 B32 ops (edges only)
      int r = i / TCOLS;
      int c = i % TCOLS;
      int gy = gy0 + r - 1;     gy = gy < 0 ? 0 : (gy >= H ? H - 1 : gy);
      int gx = gx0 + c - TPADL; gx = gx < 0 ? 0 : (gx >= W ? W - 1 : gx);
      unsigned goff = (unsigned)(((size_t)gy * (size_t)W + (size_t)gx) * 4u);
      unsigned loff = (unsigned)(uintptr_t)(&tile[i]);
      asm volatile("global_load_async_to_lds_b32 %0, %1, %2"
                   :: "v"(loff), "v"(goff), "s"(src) : "memory");
    }
  }
}

__device__ __forceinline__ void wait_async() {
  asm volatile("s_wait_asynccnt 0" ::: "memory");
}

__device__ __forceinline__ bool fast_block(int bx, int by, int H, int W) {
  int gx0 = bx * TW;
  int gy0 = by * TH;
  return (gx0 >= TPADL) && (gx0 + TW + TPADL <= W) && (gy0 >= 1) && (gy0 + TH + 1 <= H);
}

// ---------------- NMS score (shared by hist + compact passes) ----------------
// tile center for thread (tx,ty) is tile[(ty+1)*TCOLS + tx + TPADL]
__device__ __forceinline__ float nms_score(float lo, float hi, const float* tile,
                                           int tx, int ty, int gx, int gy, int H, int W) {
  float c = tile[(ty + 1) * TCOLS + (tx + TPADL)];
  float m = -INFINITY;
#pragma unroll
  for (int dy = 0; dy < 3; ++dy)
#pragma unroll
    for (int dx = 0; dx < 3; ++dx)
      m = fmaxf(m, tile[(ty + dy) * TCOLS + (tx + TPADL - 1 + dx)]);
  float v = (c - m + NMS_EPS_C > 0.0f) ? c : 0.0f;
  if (!(c > lo && c > hi)) v = 0.0f;
  if (gx < BORDER_C || gx >= W - BORDER_C || gy < BORDER_C || gy >= H - BORDER_C) v = 0.0f;
  return v;
}

// ---------------- kernel 0: zero hist + counter ----------------
__global__ void k_init(unsigned* __restrict__ hist, unsigned* __restrict__ cnt) {
  int i = blockIdx.x * 256 + threadIdx.x;
  if (i < NBINS) hist[i] = 0u;
  if (i == 0) *cnt = 0u;
}

// ---------------- kernel 1: streaming NMS + histogram ----------------
__global__ __launch_bounds__(TW * TH) void k_nms_hist(const float* __restrict__ low,
                                                      const float* __restrict__ cur,
                                                      const float* __restrict__ high,
                                                      unsigned* __restrict__ hist,
                                                      int H, int W) {
  __shared__ __align__(16) float tile[TSIZE];
  int tid = threadIdx.y * TW + threadIdx.x;
  bool fast = fast_block(blockIdx.x, blockIdx.y, H, W);
  issue_tile_async(tile, cur, blockIdx.x, blockIdx.y, H, W, tid, fast);

  int gx = blockIdx.x * TW + threadIdx.x;
  int gy = blockIdx.y * TH + threadIdx.y;
  bool valid = (gx < W) && (gy < H);
  size_t gi = (size_t)(valid ? gy : 0) * (size_t)W + (size_t)(valid ? gx : 0);
  float lo = low[gi];   // overlap with async tile fill
  float hi = high[gi];

  wait_async();
  __syncthreads();

  if (!valid) return;
  float v = nms_score(lo, hi, tile, threadIdx.x, threadIdx.y, gx, gy, H, W);
  if (v > 0.0f) {
    int bin = (int)(v * (float)NBINS);
    bin = bin < 0 ? 0 : (bin >= NBINS ? NBINS - 1 : bin);
    atomicAdd(&hist[bin], 1u);
  }
}

// ---------------- kernel 2: suffix-scan -> threshold ----------------
__global__ void k_scan(const unsigned* __restrict__ hist, float* __restrict__ thr, int k) {
  if (threadIdx.x == 0 && blockIdx.x == 0) {
    unsigned acc = 0u;
    int b = NBINS;
    float t = 0.0f;
    while (b > 1) {
      --b;
      acc += hist[b];
      if (acc >= (unsigned)k) { t = (float)b / (float)NBINS; break; }
    }
    thr[0] = (acc >= (unsigned)k) ? t : 0.0f;
  }
}

// ---------------- kernel 3: compaction (recompute NMS, L2-hot) ----------------
__global__ __launch_bounds__(TW * TH) void k_compact(const float* __restrict__ low,
                                                     const float* __restrict__ cur,
                                                     const float* __restrict__ high,
                                                     const float* __restrict__ thr,
                                                     unsigned* __restrict__ cnt,
                                                     float* __restrict__ cv,
                                                     int* __restrict__ ci,
                                                     int H, int W) {
  __shared__ __align__(16) float tile[TSIZE];
  int tid = threadIdx.y * TW + threadIdx.x;
  bool fast = fast_block(blockIdx.x, blockIdx.y, H, W);
  issue_tile_async(tile, cur, blockIdx.x, blockIdx.y, H, W, tid, fast);

  int gx = blockIdx.x * TW + threadIdx.x;
  int gy = blockIdx.y * TH + threadIdx.y;
  bool valid = (gx < W) && (gy < H);
  size_t gi = (size_t)(valid ? gy : 0) * (size_t)W + (size_t)(valid ? gx : 0);
  float lo = low[gi];
  float hi = high[gi];
  float t = thr[0];

  wait_async();
  __syncthreads();

  if (!valid) return;
  float v = nms_score(lo, hi, tile, threadIdx.x, threadIdx.y, gx, gy, H, W);
  if (v > 0.0f && v >= t) {
    unsigned p = atomicAdd(cnt, 1u);
    if (p < (unsigned)CAP) {
      cv[p] = v;
      ci[p] = gy * W + gx;
    }
  }
}

// ---------------- kernel 4: one-block bitonic sort in LDS + topk_val ----------------
__global__ __launch_bounds__(1024) void k_sort(const float* __restrict__ cv,
                                               const int* __restrict__ ci,
                                               const unsigned* __restrict__ cnt,
                                               int* __restrict__ sel,
                                               float* __restrict__ out_val,
                                               int k) {
  __shared__ float sv[CAP];
  __shared__ int   si[CAP];
  unsigned c0 = *cnt;
  int n = (int)(c0 < (unsigned)CAP ? c0 : (unsigned)CAP);
  for (int i = threadIdx.x; i < CAP; i += 1024) {
    if (i < n) { sv[i] = cv[i]; si[i] = ci[i]; }
    else       { sv[i] = -1.0f; si[i] = 0x7FFFFFFF; }
  }
  __syncthreads();
  // bitonic sort: descending value, ascending index on ties (matches lax.top_k)
  for (unsigned size = 2; size <= (unsigned)CAP; size <<= 1) {
    for (unsigned stride = size >> 1; stride > 0; stride >>= 1) {
      for (unsigned t = threadIdx.x; t < (unsigned)CAP; t += 1024) {
        unsigned j = t ^ stride;
        if (j > t) {
          float va = sv[t], vb = sv[j];
          int   ia = si[t], ib = si[j];
          bool aFirst = (va > vb) || (va == vb && ia <= ib);
          bool up = (t & size) == 0;
          if (up ? !aFirst : aFirst) {
            sv[t] = vb; sv[j] = va;
            si[t] = ib; si[j] = ia;
          }
        }
      }
      __syncthreads();
    }
  }
  for (int i = threadIdx.x; i < k; i += 1024) {
    float v = (i < n) ? sv[i] : 0.0f;
    int  id = (i < n) ? si[i] : i;   // pad like top_k over mostly-zero tail
    out_val[i] = v;
    sel[i] = id;
  }
}

// ---------------- kernel 5: WMMA batched 27-tap conv refinement ----------------
// One wave handles 16 candidates. A(16x32 f16) = candidate neighborhoods
// (K = z*9 + (dy+1)*3 + (dx+1), K=27..31 zero-padded). B(32x16 f16) columns:
// N=0: ones (den), N=1: z-1 (Wz), N=2: dy (Wy), N=3: dx (Wx), rest zero.
// D(16x16 f32) -> transpose out via LDS. Layouts per CDNA5 ISA 7.12.2.
__global__ __launch_bounds__(32) void k_refine_wmma(const float* __restrict__ low,
                                                    const float* __restrict__ cur,
                                                    const float* __restrict__ high,
                                                    const int* __restrict__ sel,
                                                    float* __restrict__ out,
                                                    int H, int W, int k) {
  __shared__ float xch[32 * 8];
  int lane = threadIdx.x;           // 0..31
  int m = lane & 15;                // A row / D row group; also B column
  bool hiHalf = lane >= 16;
  int cand = blockIdx.x * 16 + m;
  int idx = (cand < k) ? sel[cand] : 0;
  int cy = idx / W;
  int cx = idx - cy * W;

  v16h a, b;
#pragma unroll
  for (int s = 0; s < 16; ++s) {
    // A-matrix 16x32 f16 layout: lanes 0-15 hold K {0..7,16..23}, lanes 16-31 hold K {8..15,24..31}
    int Ka = (s < 8) ? s : s + 8;
    if (hiHalf) Ka += 8;
    float av = 0.0f;
    if (Ka < 27) {
      int z = Ka / 9, rem = Ka % 9;
      int dy = rem / 3 - 1, dx = rem % 3 - 1;
      int yy = cy + dy, xx = cx + dx;
      if (yy >= 0 && yy < H && xx >= 0 && xx < W) {
        const float* src = (z == 0) ? low : ((z == 1) ? cur : high);
        av = src[(size_t)yy * (size_t)W + (size_t)xx];
      }
    }
    a[s] = (_Float16)av;

    // B-matrix 32x16 f16 layout: lanes 0-15 hold K {0..7,16..23}, lanes 16-31 hold K {8..15,24..31}
    int Kb = ((s < 8) ? s : s + 8) + (hiHalf ? 8 : 0);
    float bw = 0.0f;
    if (Kb < 27) {
      int z = Kb / 9, rem = Kb % 9;
      int dy = rem / 3 - 1, dx = rem % 3 - 1;
      int n = m;
      if (n == 0)      bw = 1.0f;
      else if (n == 1) bw = (float)(z - 1);
      else if (n == 2) bw = (float)dy;
      else if (n == 3) bw = (float)dx;
    }
    b[s] = (_Float16)bw;
  }

  v8f c = {};
  c = __builtin_amdgcn_wmma_f32_16x16x32_f16(false, a, false, b, (short)0, c, false, false);

#pragma unroll
  for (int e = 0; e < 8; ++e) xch[lane * 8 + e] = c[e];
  __syncthreads();

  if (lane < 16 && cand < k) {
    // D[M][N]: M<8 -> lane=N, elem=M ; M>=8 -> lane=N+16, elem=M-8
    int e  = m & 7;
    int lb = (m >= 8) ? 16 : 0;
    float den = xch[(lb + 0) * 8 + e] + SAM_EPS_C;
    float nz  = xch[(lb + 1) * 8 + e];
    float ny  = xch[(lb + 2) * 8 + e];
    float nx  = xch[(lb + 3) * 8 + e];
    float sc   = nz / den;
    float ysub = ((float)cy + ny / den) / (float)H;
    float xsub = ((float)cx + nx / den) / (float)W;
    int mn = H < W ? H : W;
    float aa = sc * (1.0f / (float)mn);   // MR_SIZE / min_size * s
    float* fa = out + k + (size_t)cand * 6;
    fa[0] = aa;   fa[1] = 0.0f; fa[2] = xsub;
    fa[3] = 0.0f; fa[4] = aa;   fa[5] = ysub;
  }
}

// ---------------- host launcher ----------------
extern "C" void kernel_launch(void* const* d_in, const int* in_sizes, int n_in,
                              void* d_out, int out_size, void* d_ws, size_t ws_size,
                              hipStream_t stream) {
  const float* low  = (const float*)d_in[0];
  const float* cur  = (const float*)d_in[1];
  const float* high = (const float*)d_in[2];

  int HW = in_sizes[0];
  int W = (int)(sqrtf((float)HW) + 0.5f);
  int H = HW / W;
  int k = out_size / 7;   // out = topk_val(k) + full_A(k*6)

  char* ws = (char*)d_ws;
  unsigned* hist = (unsigned*)(ws);                       // 8192 * 4 = 32768 B
  unsigned* cnt  = (unsigned*)(ws + 32768);               // 4 B
  float*    thr  = (float*)  (ws + 32772);                // 4 B
  int*      sel  = (int*)    (ws + 32800);                // k ints (<= 32768 B)
  float*    cv   = (float*)  (ws + 32800 + 32768);        // CAP floats
  int*      ci   = (int*)    (ws + 32800 + 32768 + 4 * CAP); // CAP ints

  float* out = (float*)d_out;

  dim3 blk(TW, TH);
  dim3 grd((W + TW - 1) / TW, (H + TH - 1) / TH);

  k_init<<<(NBINS + 255) / 256, 256, 0, stream>>>(hist, cnt);
  k_nms_hist<<<grd, blk, 0, stream>>>(low, cur, high, hist, H, W);
  k_scan<<<1, 1, 0, stream>>>(hist, thr, k);
  k_compact<<<grd, blk, 0, stream>>>(low, cur, high, thr, cnt, cv, ci, H, W);
  k_sort<<<1, 1024, 0, stream>>>(cv, ci, cnt, sel, out, k);
  k_refine_wmma<<<(k + 15) / 16, 32, 0, stream>>>(low, cur, high, sel, out, H, W, k);
}
